// LocalScopeSelfAttention_41824391528951
// MI455X (gfx1250) — matrix operand order repaired and checked by reading, not compiled, not run on hardware
//
#include <hip/hip_runtime.h>

// ---------------- CDNA5 WMMA types ----------------
typedef __attribute__((ext_vector_type(16))) __bf16 v16bf;
typedef __attribute__((ext_vector_type(8)))  __bf16 v8bf;
typedef __attribute__((ext_vector_type(4)))  __bf16 v4bf;
typedef __attribute__((ext_vector_type(8)))  float  v8f;

#define DMODEL 384
#define ND3    1152
#define NHEAD  12
#define HDIM   32
#define NTOK   49
#define MPAD   64
#define WSZ7   7

// LDS layout (bf16 element offsets)
#define OFF_XN 0                        // 64x384 (48KB), reused as attn-out
#define OFF_Q  (OFF_XN + MPAD*DMODEL)   // 64x384 row-major
#define OFF_K  (OFF_Q  + MPAD*DMODEL)   // 64x384 row-major (k contiguous)
#define OFF_VT (OFF_K  + MPAD*DMODEL)   // 384x64 TRANSPOSED (token contiguous)
#define OFF_W0 (OFF_VT + MPAD*DMODEL)   // weight tile buf0, transposed [n][32]
#define OFF_W1 (OFF_W0 + 256*32)        // weight tile buf1
#define OFF_P  (OFF_W1 + 256*32)        // 8 waves x 16x64 (16KB)
#define SMEM_ELEMS (OFF_P + 8*16*64)
#define SMEM_BYTES (SMEM_ELEMS * 2)

static __device__ __forceinline__ v8f z8() {
  v8f z;
#pragma unroll
  for (int i = 0; i < 8; ++i) z[i] = 0.f;
  return z;
}

static __device__ __forceinline__ v8f wmma_bf16(v16bf a, v16bf b, v8f c) {
  return __builtin_amdgcn_wmma_f32_16x16x32_bf16(
      false, a, false, b, (short)0, c, false, false);
}

// A-fragment (16x32 bf16, MxK) from row-major [row*ld + k0 + k]
// lanes0-15 M=lane; VGPR v: K = hl*16 + v*2  -> 2x ds_load_b128
static __device__ __forceinline__ v16bf load_a_frag(const __bf16* p, int row,
                                                    int ld, int k0, int hl) {
  const __bf16* rp = p + row * ld + k0;
  v16bf f;
#pragma unroll
  for (int v = 0; v < 8; ++v) {
    int kb = hl * 16 + v * 2;
    f[2 * v]     = rp[kb];
    f[2 * v + 1] = rp[kb + 1];
  }
  return f;
}

// B-fragment (32x16, KxN) where K is contiguous: B[k][n] = p[n*ld + koff + k]
// lane reads 16 contiguous bf16 -> 2x ds_load_b128
static __device__ __forceinline__ v16bf load_b_kcontig(const __bf16* p, int col,
                                                       int ld, int koff, int hl) {
  const __bf16* rp = p + col * ld + koff + hl * 16;
  v16bf f;
#pragma unroll
  for (int v = 0; v < 8; ++v) {
    f[2 * v]     = rp[v * 2];
    f[2 * v + 1] = rp[v * 2 + 1];
  }
  return f;
}

// Stage one 32 x WT f32 slab of W into a TRANSPOSED bf16 tile [n][32].
// Lane-coalesced b32 global loads, packed b64 LDS stores.
template <int NT>
static __device__ __forceinline__ void stage_slab(
    __bf16* __restrict__ dst, const float* __restrict__ gw, int gld,
    int n0, int k0, int tid) {
  constexpr int WT = NT * 128;
#pragma unroll
  for (int ii = 0; ii < NT * 4; ++ii) {
    int idx = tid + ii * 256;          // WT*8 groups of 4 k-values
    int n   = idx % WT;
    int kg  = idx / WT;
    const float* gp = gw + (size_t)(k0 + kg * 4) * gld + n0 + n;
    float f0 = gp[0];
    float f1 = gp[(size_t)gld];
    float f2 = gp[(size_t)2 * gld];
    float f3 = gp[(size_t)3 * gld];
    v4bf pk;
    pk[0] = (__bf16)f0; pk[1] = (__bf16)f1;
    pk[2] = (__bf16)f2; pk[3] = (__bf16)f3;
    *(v4bf*)&dst[n * 32 + kg * 4] = pk;
  }
}

// One 64 x (NT*128) GEMM macro-tile with DOUBLE-BUFFERED weight staging:
// one barrier per k-step; next slab's global loads overlap current WMMAs.
template <int NT>
static __device__ __forceinline__ void gemm_block(
    const __bf16* __restrict__ s_a, __bf16* __restrict__ s_w0,
    __bf16* __restrict__ s_w1, const float* __restrict__ gw, int gld, int n0,
    int tid, int wv, int l16, int hl, v8f (&acc)[NT][4]) {
  stage_slab<NT>(s_w0, gw, gld, n0, 0, tid);
  for (int ks = 0; ks < 12; ++ks) {
    __syncthreads();
    __bf16* cur = (ks & 1) ? s_w1 : s_w0;
    __bf16* nxt = (ks & 1) ? s_w0 : s_w1;
    if (ks < 11) {
      stage_slab<NT>(nxt, gw, gld, n0, (ks + 1) * 32, tid);
      if (ks < 10)  // global_prefetch_b8 two slabs ahead
        __builtin_prefetch(&gw[(size_t)(ks + 2) * 32 * gld + n0 + tid], 0, 1);
    }
    const int k0 = ks * 32;
    v16bf fb[NT];
#pragma unroll
    for (int t = 0; t < NT; ++t)
      fb[t] = load_b_kcontig(cur, (wv + t * 8) * 16 + l16, 32, 0, hl);
#pragma unroll
    for (int mi = 0; mi < 4; ++mi) {
      v16bf fa = load_a_frag(s_a, mi * 16 + l16, DMODEL, k0, hl);
#pragma unroll
      for (int t = 0; t < NT; ++t)
        acc[t][mi] = wmma_bf16(fa, fb[t], acc[t][mi]);
    }
  }
}

// QKV epilogue: +bias, q*scale, scatter into Q/K row-major or V transposed
static __device__ __forceinline__ void qkv_store(const v8f (&a)[4], int ng,
                                                 const float* __restrict__ qkv_b,
                                                 __bf16* s_q, __bf16* s_k,
                                                 __bf16* s_vt, int hl) {
  const float bb = qkv_b[ng];
  if (ng < 384) {
    const float qs = 0.17677669529663687f;  // hd^-0.5
#pragma unroll
    for (int mi = 0; mi < 4; ++mi)
#pragma unroll
      for (int r = 0; r < 8; ++r)
        s_q[(mi * 16 + r + hl * 8) * DMODEL + ng] = (__bf16)((a[mi][r] + bb) * qs);
  } else if (ng < 768) {
    const int col = ng - 384;
#pragma unroll
    for (int mi = 0; mi < 4; ++mi)
#pragma unroll
      for (int r = 0; r < 8; ++r)
        s_k[(mi * 16 + r + hl * 8) * DMODEL + col] = (__bf16)(a[mi][r] + bb);
  } else {
    const int col = ng - 768;   // transposed: [d][token], contiguous b128
#pragma unroll
    for (int mi = 0; mi < 4; ++mi) {
      v8bf pk;
#pragma unroll
      for (int r = 0; r < 8; ++r) pk[r] = (__bf16)(a[mi][r] + bb);
      *(v8bf*)&s_vt[col * MPAD + mi * 16 + hl * 8] = pk;
    }
  }
}

extern "C" __global__ __launch_bounds__(256, 1)
void swin_window_attn_kernel(const float* __restrict__ x,
                             const float* __restrict__ ln_g,
                             const float* __restrict__ ln_b,
                             const float* __restrict__ qkv_w,
                             const float* __restrict__ qkv_b,
                             const float* __restrict__ proj_w,
                             const float* __restrict__ proj_b,
                             const float* __restrict__ rel_tab,
                             float* __restrict__ out) {
  extern __shared__ __align__(16) char smem_raw[];
  __bf16* smem  = (__bf16*)smem_raw;
  __bf16* s_xn  = smem + OFF_XN;  // normalized x / later attn-out
  __bf16* s_q   = smem + OFF_Q;
  __bf16* s_k   = smem + OFF_K;
  __bf16* s_vt  = smem + OFF_VT;
  __bf16* s_w0  = smem + OFF_W0;
  __bf16* s_w1  = smem + OFF_W1;
  __bf16* s_p   = smem + OFF_P;

  const int tid  = threadIdx.x;
  const int wv   = tid >> 5;
  const int lane = tid & 31;
  const int hl   = lane >> 4;
  const int l16  = lane & 15;

  const int win = blockIdx.x;       // 24 images x 8 x 8 windows
  const int bt  = win >> 6;
  const int wy  = (win & 63) >> 3;
  const int wx  = win & 7;

  // ---------------- Phase 0: load + LayerNorm -> s_xn (bf16) -------------
  const int c0 = lane * 12;
  float gam[12], bet[12];
  {
    const float4* g4 = (const float4*)(ln_g + c0);
    const float4* b4 = (const float4*)(ln_b + c0);
    float4 g0 = g4[0], g1 = g4[1], g2 = g4[2];
    float4 b0 = b4[0], b1 = b4[1], b2 = b4[2];
    gam[0]=g0.x; gam[1]=g0.y; gam[2]=g0.z; gam[3]=g0.w;
    gam[4]=g1.x; gam[5]=g1.y; gam[6]=g1.z; gam[7]=g1.w;
    gam[8]=g2.x; gam[9]=g2.y; gam[10]=g2.z; gam[11]=g2.w;
    bet[0]=b0.x; bet[1]=b0.y; bet[2]=b0.z; bet[3]=b0.w;
    bet[4]=b1.x; bet[5]=b1.y; bet[6]=b1.z; bet[7]=b1.w;
    bet[8]=b2.x; bet[9]=b2.y; bet[10]=b2.z; bet[11]=b2.w;
  }
  for (int m = wv; m < NTOK; m += 8) {
    const int i = m / WSZ7, j = m % WSZ7;
    const size_t grow = (size_t)bt * 3136 + (size_t)(wy * 7 + i) * 56 + (wx * 7 + j);
    const float* rp = x + grow * DMODEL + c0;
    float vr[12];
    const float4* r4 = (const float4*)rp;
    float4 a0 = r4[0], a1 = r4[1], a2 = r4[2];
    vr[0]=a0.x; vr[1]=a0.y; vr[2]=a0.z; vr[3]=a0.w;
    vr[4]=a1.x; vr[5]=a1.y; vr[6]=a1.z; vr[7]=a1.w;
    vr[8]=a2.x; vr[9]=a2.y; vr[10]=a2.z; vr[11]=a2.w;
    float s = 0.f, s2 = 0.f;
#pragma unroll
    for (int c = 0; c < 12; ++c) { s += vr[c]; s2 += vr[c] * vr[c]; }
#pragma unroll
    for (int mk = 16; mk >= 1; mk >>= 1) {
      s  += __shfl_xor(s,  mk, 32);
      s2 += __shfl_xor(s2, mk, 32);
    }
    const float mean = s * (1.f / DMODEL);
    const float var  = s2 * (1.f / DMODEL) - mean * mean;
    const float rstd = rsqrtf(var + 1e-5f);
#pragma unroll
    for (int c = 0; c < 12; ++c) {
      float y = (vr[c] - mean) * rstd * gam[c] + bet[c];
      s_xn[m * DMODEL + c0 + c] = (__bf16)y;
    }
  }
  for (int e = tid; e < (MPAD - NTOK) * DMODEL; e += 256)
    s_xn[NTOK * DMODEL + e] = (__bf16)0.f;
  __syncthreads();

  // ---------------- Phase 1: QKV GEMM 64x384 @ 384x1152 -------------------
  // 4 chunks of 256 cols (2 n-tiles/wave) + 1 chunk of 128 (1 n-tile/wave)
  for (int c = 0; c < 4; ++c) {
    v8f acc[2][4];
#pragma unroll
    for (int t = 0; t < 2; ++t)
#pragma unroll
      for (int mi = 0; mi < 4; ++mi) acc[t][mi] = z8();
    gemm_block<2>(s_xn, s_w0, s_w1, qkv_w, ND3, c * 256, tid, wv, l16, hl, acc);
#pragma unroll
    for (int t = 0; t < 2; ++t)
      qkv_store(acc[t], c * 256 + (wv + t * 8) * 16 + l16, qkv_b, s_q, s_k, s_vt, hl);
  }
  {
    v8f acc[1][4];
#pragma unroll
    for (int mi = 0; mi < 4; ++mi) acc[0][mi] = z8();
    gemm_block<1>(s_xn, s_w0, s_w1, qkv_w, ND3, 1024, tid, wv, l16, hl, acc);
    qkv_store(acc[0], 1024 + wv * 16 + l16, qkv_b, s_q, s_k, s_vt, hl);
  }
  __syncthreads();

  // ---------------- Phase 2: per-head attention ---------------------------
  for (int h = wv; h < NHEAD; h += 8) {
    // K^T fragments: B[k=headdim][n=token], k contiguous in s_k rows
    v16bf fbK[4];
#pragma unroll
    for (int ni = 0; ni < 4; ++ni)
      fbK[ni] = load_b_kcontig(s_k, ni * 16 + l16, DMODEL, h * HDIM, hl);
    // V fragments: B[k=token][n=headdim]; tokens contiguous in s_vt rows
    v16bf fbV[2][2];
#pragma unroll
    for (int ks = 0; ks < 2; ++ks)
#pragma unroll
      for (int ni = 0; ni < 2; ++ni)
        fbV[ks][ni] = load_b_kcontig(s_vt, h * HDIM + ni * 16 + l16, MPAD,
                                     ks * 32, hl);

    for (int mi = 0; mi < 4; ++mi) {
      // S = (q*scale) @ K^T : 16 rows x 64 cols
      v8f S[4];
#pragma unroll
      for (int ni = 0; ni < 4; ++ni) S[ni] = z8();
      v16bf fa = load_a_frag(s_q, mi * 16 + l16, DMODEL, h * HDIM, hl);
#pragma unroll
      for (int ni = 0; ni < 4; ++ni) S[ni] = wmma_bf16(fa, fbK[ni], S[ni]);

      // bias + mask + softmax; rows are (r, lane-half), cols across 16 lanes
#pragma unroll
      for (int r = 0; r < 8; ++r) {
        const int m  = mi * 16 + r + hl * 8;
        const int im = m / WSZ7, jm = m % WSZ7;
        float vals[4];
#pragma unroll
        for (int ni = 0; ni < 4; ++ni) {
          const int n = ni * 16 + l16;
          float sv = S[ni][r];
          if (n >= NTOK || m >= NTOK) {
            sv = -1e30f;
          } else {
            const int in_ = n / WSZ7, jn = n % WSZ7;
            const int idx = (im - in_ + 6) * 13 + (jm - jn + 6);
            sv += rel_tab[idx * NHEAD + h];
          }
          vals[ni] = sv;
        }
        float mx = fmaxf(fmaxf(vals[0], vals[1]), fmaxf(vals[2], vals[3]));
#pragma unroll
        for (int mk = 1; mk < 16; mk <<= 1) mx = fmaxf(mx, __shfl_xor(mx, mk, 32));
        float sum = 0.f;
#pragma unroll
        for (int ni = 0; ni < 4; ++ni) { vals[ni] = __expf(vals[ni] - mx); sum += vals[ni]; }
#pragma unroll
        for (int mk = 1; mk < 16; mk <<= 1) sum += __shfl_xor(sum, mk, 32);
        const float inv = 1.f / sum;
#pragma unroll
        for (int ni = 0; ni < 4; ++ni)
          s_p[wv * 1024 + (r + hl * 8) * 64 + ni * 16 + l16] = (__bf16)(vals[ni] * inv);
      }

      // O = P(16x64) @ V_h(64x32)
      v8f O[2];
#pragma unroll
      for (int ni = 0; ni < 2; ++ni) O[ni] = z8();
#pragma unroll
      for (int ks = 0; ks < 2; ++ks) {
        v16bf fp = load_a_frag(s_p + wv * 1024, l16, 64, ks * 32, hl);
#pragma unroll
        for (int ni = 0; ni < 2; ++ni) O[ni] = wmma_bf16(fp, fbV[ks][ni], O[ni]);
      }
#pragma unroll
      for (int ni = 0; ni < 2; ++ni)
#pragma unroll
        for (int r = 0; r < 8; ++r) {
          int m = mi * 16 + r + hl * 8;
          s_xn[m * DMODEL + h * HDIM + ni * 16 + l16] = (__bf16)O[ni][r];
        }
    }
  }
  __syncthreads();

  // ---------------- Phase 3: projection 64x384 @ 384x384 ------------------
  // chunk0: 256 cols (2 tiles/wave), chunk1: 128 cols (1 tile/wave)
  {
    v8f acc[2][4];
#pragma unroll
    for (int t = 0; t < 2; ++t)
#pragma unroll
      for (int mi = 0; mi < 4; ++mi) acc[t][mi] = z8();
    gemm_block<2>(s_xn, s_w0, s_w1, proj_w, DMODEL, 0, tid, wv, l16, hl, acc);
#pragma unroll
    for (int t = 0; t < 2; ++t) {
      const int ng = (wv + t * 8) * 16 + l16;
      const float bb = proj_b[ng];
#pragma unroll
      for (int mi = 0; mi < 4; ++mi)
#pragma unroll
        for (int r = 0; r < 8; ++r) {
          const int m = mi * 16 + r + hl * 8;
          if (m < NTOK) {
            const int i = m / WSZ7, j = m % WSZ7;
            const size_t grow =
                (size_t)bt * 3136 + (size_t)(wy * 7 + i) * 56 + (wx * 7 + j);
            out[grow * DMODEL + ng] = acc[t][mi][r] + bb;
          }
        }
    }
  }
  {
    v8f acc[1][4];
#pragma unroll
    for (int mi = 0; mi < 4; ++mi) acc[0][mi] = z8();
    gemm_block<1>(s_xn, s_w0, s_w1, proj_w, DMODEL, 256, tid, wv, l16, hl, acc);
    const int ng = 256 + wv * 16 + l16;
    const float bb = proj_b[ng];
#pragma unroll
    for (int mi = 0; mi < 4; ++mi)
#pragma unroll
      for (int r = 0; r < 8; ++r) {
        const int m = mi * 16 + r + hl * 8;
        if (m < NTOK) {
          const int i = m / WSZ7, j = m % WSZ7;
          const size_t grow =
              (size_t)bt * 3136 + (size_t)(wy * 7 + i) * 56 + (wx * 7 + j);
          out[grow * DMODEL + ng] = acc[0][mi][r] + bb;
        }
      }
  }
}

extern "C" void kernel_launch(void* const* d_in, const int* in_sizes, int n_in,
                              void* d_out, int out_size, void* d_ws, size_t ws_size,
                              hipStream_t stream) {
  const float* x       = (const float*)d_in[0];
  const float* ln_g    = (const float*)d_in[1];
  const float* ln_b    = (const float*)d_in[2];
  const float* qkv_w   = (const float*)d_in[3];
  const float* qkv_b   = (const float*)d_in[4];
  const float* proj_w  = (const float*)d_in[5];
  const float* proj_b  = (const float*)d_in[6];
  const float* rel_tab = (const float*)d_in[7];
  float* out = (float*)d_out;

  (void)in_sizes; (void)n_in; (void)out_size; (void)d_ws; (void)ws_size;

  static_assert(SMEM_BYTES <= 320 * 1024, "LDS budget exceeded");
  (void)hipFuncSetAttribute((const void*)swin_window_attn_kernel,
                            hipFuncAttributeMaxDynamicSharedMemorySize,
                            SMEM_BYTES);

  swin_window_attn_kernel<<<dim3(1536), dim3(256), SMEM_BYTES, stream>>>(
      x, ln_g, ln_b, qkv_w, qkv_b, proj_w, proj_b, rel_tab, out);
}
